// VAE_35897336660302
// MI455X (gfx1250) — matrix operand order repaired and checked by reading, not compiled, not run on hardware
//
#include <hip/hip_runtime.h>
#include <hip/hip_bf16.h>
#include <math.h>

#define NMAX 200
#define BATCH 512
#define LATENTD 128
#define ENCH 256
#define DECH 512

typedef __attribute__((ext_vector_type(16))) __bf16 v16bf;
typedef __attribute__((ext_vector_type(8)))  float  v8f;

#if defined(__gfx1250__) && __has_builtin(__builtin_amdgcn_global_load_async_to_lds_b32) && __has_builtin(__builtin_amdgcn_s_wait_asynccnt)
#define USE_ASYNC 1
typedef __attribute__((address_space(1))) int gas_int;
typedef __attribute__((address_space(3))) int las_int;
#else
#define USE_ASYNC 0
#endif

__device__ __forceinline__ float sigf(float x) { return 1.0f / (1.0f + expf(-x)); }

// ---------------- init / conversion kernels ----------------

__global__ void k_init_out(float* out) {
  int i = blockIdx.x * blockDim.x + threadIdx.x;
  if (i < BATCH * 5) out[(size_t)(NMAX - 1) * (BATCH * 5) + i] = 0.0f;  // strokes row 199 = 0
  if (i == 0) { out[512000] = 0.0f; out[512001] = 0.0f; }               // offset_loss, pen_loss
}

__global__ void k_f32_to_bf16(const float* __restrict__ s, __bf16* __restrict__ d, int n) {
  int i = blockIdx.x * blockDim.x + threadIdx.x;
  if (i < n) d[i] = (__bf16)s[i];
}

// dec_Wih (133 x 2048) -> padded (160 x 2048), zero rows 133..159
__global__ void k_pad_wih(const float* __restrict__ s, __bf16* __restrict__ d) {
  int i = blockIdx.x * blockDim.x + threadIdx.x;
  if (i < 160 * 2048) {
    int r = i >> 11;
    d[i] = (r < 133) ? (__bf16)s[i] : (__bf16)0.0f;
  }
}

// W_out (512 x 123) -> padded (512 x 128), zero cols 123..127
__global__ void k_pad_wout(const float* __restrict__ s, __bf16* __restrict__ d) {
  int i = blockIdx.x * blockDim.x + threadIdx.x;
  if (i < 512 * 128) {
    int r = i >> 7, c = i & 127;
    d[i] = (c < 123) ? (__bf16)s[r * 123 + c] : (__bf16)0.0f;
  }
}

// ---------------- WMMA operand loaders ----------------

// A: 16x32 bf16 tile from LDS row-major (stride elements)
__device__ __forceinline__ v16bf load_A_lds(const __bf16* base, int stride, int k0, int lane) {
  int row = lane & 15;
  int kb  = k0 + ((lane >> 4) << 3);
  const __bf16* p = base + row * stride + kb;
  v16bf a;
#pragma unroll
  for (int j = 0; j < 8; ++j) { a[j] = p[j]; a[8 + j] = p[16 + j]; }
  return a;
}

// B: 32x16 bf16 tile from global row-major K x N (stride elements), col base nb
__device__ __forceinline__ v16bf load_B_gl(const __bf16* base, int stride, int k0, int nb, int lane) {
  const __bf16* p = base + (size_t)(k0 + ((lane >> 4) << 3)) * stride + nb + (lane & 15);
  v16bf b;
#pragma unroll
  for (int j = 0; j < 8; ++j) { b[j] = p[(size_t)j * stride]; b[8 + j] = p[(size_t)(16 + j) * stride]; }
  return b;
}

// ---------------- encoder: per-direction LSTM over 201 steps ----------------

__global__ __launch_bounds__(256)
void k_encoder(const float* __restrict__ x,
               const float* __restrict__ Wih_f, const float* __restrict__ b_f,
               const float* __restrict__ Wih_b, const float* __restrict__ b_b,
               const __bf16* __restrict__ Whh_f, const __bf16* __restrict__ Whh_b,
               float* __restrict__ hf_out, float* __restrict__ hb_out)
{
  const int tile = blockIdx.x;     // batch tile 0..31
  const int dir  = blockIdx.y;     // 0 fwd, 1 bwd
  const int tid  = threadIdx.x;
  const int lane = tid & 31, wave = tid >> 5;

  const float* Wih  = dir ? Wih_b : Wih_f;
  const float* bias = dir ? b_b   : b_f;
  const __bf16* Whh = dir ? Whh_b : Whh_f;
  float* hout       = dir ? hb_out : hf_out;

  __shared__ float  s_gate[16][1024];  // 64 KB
  __shared__ __bf16 s_h[16][ENCH];     // 8 KB
  __shared__ float  s_x[2][16][8];     // double-buffered input tile
  __shared__ float  s_wih[5][1024];    // 20 KB staged input weights
  __shared__ float  s_bias[1024];      // 4 KB

  float c_reg[16];
#pragma unroll
  for (int r = 0; r < 16; ++r) c_reg[r] = 0.0f;
  for (int idx = tid; idx < 16 * ENCH; idx += 256) ((__bf16*)s_h)[idx] = (__bf16)0.0f;
  for (int idx = tid; idx < 5 * 1024; idx += 256) s_wih[idx >> 10][idx & 1023] = Wih[idx];
  for (int idx = tid; idx < 1024; idx += 256) s_bias[idx] = bias[idx];

#if USE_ASYNC
  if (tid < 80) {  // prime async fetch of x tile for t=0
    int b5 = tid / 5, k5 = tid % 5;
    int t0 = dir ? NMAX : 0;
    const float* gp = &x[(size_t)t0 * (BATCH * 5) + (size_t)(tile * 16 + b5) * 5 + k5];
    __builtin_amdgcn_global_load_async_to_lds_b32((gas_int*)gp, (las_int*)&s_x[0][b5][k5], 0, 0);
  }
#endif

  const int brow = tid & 15, jcol0 = tid >> 4;

  for (int t = 0; t <= NMAX; ++t) {
    __syncthreads();
    int ts = dir ? (NMAX - t) : t;
#if USE_ASYNC
    if (tid < 80) __builtin_amdgcn_s_wait_asynccnt(0);
    __syncthreads();
    if (t < NMAX && tid < 80) {  // async prefetch of next x tile into other buffer
      int b5 = tid / 5, k5 = tid % 5;
      int tn = dir ? (NMAX - (t + 1)) : (t + 1);
      const float* gp = &x[(size_t)tn * (BATCH * 5) + (size_t)(tile * 16 + b5) * 5 + k5];
      __builtin_amdgcn_global_load_async_to_lds_b32((gas_int*)gp, (las_int*)&s_x[(t + 1) & 1][b5][k5], 0, 0);
    }
    const float (*sx_cur)[8] = s_x[t & 1];
    (void)ts;
#else
    if (tid < 80) {
      int b5 = tid / 5, k5 = tid % 5;
      s_x[0][b5][k5] = x[(size_t)ts * (BATCH * 5) + (size_t)(tile * 16 + b5) * 5 + k5];
    }
    __syncthreads();
    const float (*sx_cur)[8] = s_x[0];
    (void)ts;
#endif
    // g = x @ Wih + b   (K=5, VALU, LDS-staged weights)
    for (int idx = tid; idx < 16 * 1024; idx += 256) {
      int b = idx >> 10, col = idx & 1023;
      float v = s_bias[col];
#pragma unroll
      for (int k = 0; k < 5; ++k) v += sx_cur[b][k] * s_wih[k][col];
      s_gate[b][col] = v;
    }
    __syncthreads();
    // g += h @ Whh  via WMMA bf16 (64 column tiles, K=256)
    for (int ct = wave; ct < 64; ct += 8) {
      int nb = ct * 16;
      int m0 = (lane >> 4) * 8, nn = lane & 15;
      v8f acc;
#pragma unroll
      for (int r = 0; r < 8; ++r) acc[r] = s_gate[m0 + r][nb + nn];
      for (int kk = 0; kk < 8; ++kk) {
        int k0 = kk * 32;
        __builtin_prefetch(Whh + (size_t)(k0 + 32 + lane) * 1024 + nb, 0, 1);
        v16bf a  = load_A_lds(&s_h[0][0], ENCH, k0, lane);
        v16bf bm = load_B_gl(Whh, 1024, k0, nb, lane);
        acc = __builtin_amdgcn_wmma_f32_16x16x32_bf16(false, a, false, bm, (short)0, acc, false, false);
      }
#pragma unroll
      for (int r = 0; r < 8; ++r) s_gate[m0 + r][nb + nn] = acc[r];
    }
    __syncthreads();
    // LSTM nonlinearity; each thread owns 16 (b,j) cells
#pragma unroll
    for (int r = 0; r < 16; ++r) {
      int j = jcol0 + r * 16;
      float gi = s_gate[brow][j];
      float gf = s_gate[brow][256 + j];
      float gg = s_gate[brow][512 + j];
      float go = s_gate[brow][768 + j];
      float c = sigf(gf) * c_reg[r] + sigf(gi) * tanhf(gg);
      float h = sigf(go) * tanhf(c);
      c_reg[r] = c;
      s_h[brow][j] = (__bf16)h;
      if (t == NMAX) hout[(size_t)(tile * 16 + brow) * ENCH + j] = h;
    }
  }
}

// ---------------- latent stage (small; VALU) ----------------

__global__ __launch_bounds__(128)
void k_latent(const float* __restrict__ hf, const float* __restrict__ hb,
              const float* __restrict__ eps,
              const float* __restrict__ W_mu, const float* __restrict__ b_mu,
              const float* __restrict__ W_lv, const float* __restrict__ b_lv,
              const float* __restrict__ W_fc, const float* __restrict__ b_fc,
              float* __restrict__ out_mean, float* __restrict__ out_logvar,
              float* __restrict__ z_out, float* __restrict__ h0_out, float* __restrict__ c0_out)
{
  int b = blockIdx.x, l = threadIdx.x;
  __shared__ float s_he[2 * ENCH];
  __shared__ float s_z[LATENTD];
#pragma unroll
  for (int r = 0; r < 4; ++r) {
    int idx = l + r * 128;
    s_he[idx] = (idx < ENCH) ? hf[(size_t)b * ENCH + idx] : hb[(size_t)b * ENCH + idx - ENCH];
  }
  __syncthreads();
  float m = b_mu[l], lv = b_lv[l];
  for (int k = 0; k < 2 * ENCH; ++k) {
    float h = s_he[k];
    m  += h * W_mu[k * LATENTD + l];
    lv += h * W_lv[k * LATENTD + l];
  }
  out_mean[(size_t)b * LATENTD + l]   = m;
  out_logvar[(size_t)b * LATENTD + l] = lv;
  float z = m + expf(lv * 0.5f) * eps[(size_t)b * LATENTD + l];
  z_out[(size_t)b * LATENTD + l] = z;
  s_z[l] = z;
  __syncthreads();
#pragma unroll
  for (int r = 0; r < 8; ++r) {
    int col = l + r * 128;
    float v = b_fc[col];
    for (int k = 0; k < LATENTD; ++k) v += s_z[k] * W_fc[k * 1024 + col];
    v = tanhf(v);
    if (col < DECH) h0_out[(size_t)b * DECH + col]        = v;
    else            c0_out[(size_t)b * DECH + col - DECH] = v;
  }
}

// ---------------- decoder: 199-step LSTM + GMM head + losses ----------------

__global__ __launch_bounds__(256)
void k_decoder(const float* __restrict__ x, const int* __restrict__ N_s,
               const float* __restrict__ z, const float* __restrict__ h0,
               const float* __restrict__ c0,
               const __bf16* __restrict__ Wih,   // 160 x 2048 (padded)
               const __bf16* __restrict__ Whh,   // 512 x 2048
               const float* __restrict__ dec_b,
               const __bf16* __restrict__ Wout,  // 512 x 128 (padded)
               const float* __restrict__ b_out,
               float* __restrict__ out)
{
  const int tile = blockIdx.x;
  const int tid  = threadIdx.x;
  const int lane = tid & 31, wave = tid >> 5;

  __shared__ float  s_gate[16][2048];  // 128 KB (CDNA5 320KB LDS)
  __shared__ __bf16 s_h[16][DECH];     // 16 KB
  __shared__ __bf16 s_in[16][160];     // 5 KB  (prev stroke + z, padded)
  __shared__ float  s_y[16][128];      // 8 KB
  __shared__ float  s_b[2048];         // 8 KB staged gate bias

  const int brow = tid & 15, jcol0 = tid >> 4;
  float c_reg[32];
#pragma unroll
  for (int r = 0; r < 32; ++r) {
    int j = jcol0 + r * 16;
    size_t gi = (size_t)(tile * 16 + brow) * DECH + j;
    c_reg[r] = c0[gi];
    s_h[brow][j] = (__bf16)h0[gi];
  }
  for (int idx = tid; idx < 2048; idx += 256) s_b[idx] = dec_b[idx];
  for (int idx = tid; idx < 16 * 160; idx += 256) {
    int b = idx / 160, col = idx % 160;
    float v;
    if (col < 5)        v = (col == 2) ? 1.0f : 0.0f;               // stroke0
    else if (col < 133) v = z[(size_t)(tile * 16 + b) * LATENTD + (col - 5)];
    else                v = 0.0f;
    s_in[b][col] = (__bf16)v;
  }

  float pl_local = 0.0f, ol_local = 0.0f;
  const float inv_norm = 1.0f / (512.0f * 200.0f);

  for (int t = 0; t < NMAX - 1; ++t) {   // steps i = 1..199
    __syncthreads();
    // gates = [prev,z] @ Wih + h @ Whh + b   (128 column tiles, K=160 + K=512)
    for (int ct = wave; ct < 128; ct += 8) {
      int nb = ct * 16;
      int m0 = (lane >> 4) * 8, nn = lane & 15;
      float bv = s_b[nb + nn];
      v8f acc;
#pragma unroll
      for (int r = 0; r < 8; ++r) acc[r] = bv;
      for (int kk = 0; kk < 5; ++kk) {
        int k0 = kk * 32;
        v16bf a  = load_A_lds(&s_in[0][0], 160, k0, lane);
        v16bf bm = load_B_gl(Wih, 2048, k0, nb, lane);
        acc = __builtin_amdgcn_wmma_f32_16x16x32_bf16(false, a, false, bm, (short)0, acc, false, false);
      }
      for (int kk = 0; kk < 16; ++kk) {
        int k0 = kk * 32;
        __builtin_prefetch(Whh + (size_t)(k0 + 32 + lane) * 2048 + nb, 0, 1);
        v16bf a  = load_A_lds(&s_h[0][0], DECH, k0, lane);
        v16bf bm = load_B_gl(Whh, 2048, k0, nb, lane);
        acc = __builtin_amdgcn_wmma_f32_16x16x32_bf16(false, a, false, bm, (short)0, acc, false, false);
      }
#pragma unroll
      for (int r = 0; r < 8; ++r) s_gate[m0 + r][nb + nn] = acc[r];
    }
    __syncthreads();
    // LSTM nonlinearity
#pragma unroll
    for (int r = 0; r < 32; ++r) {
      int j = jcol0 + r * 16;
      float gi = s_gate[brow][j],        gf = s_gate[brow][512 + j];
      float gg = s_gate[brow][1024 + j], go = s_gate[brow][1536 + j];
      float c = sigf(gf) * c_reg[r] + sigf(gi) * tanhf(gg);
      float h = sigf(go) * tanhf(c);
      c_reg[r] = c;
      s_h[brow][j] = (__bf16)h;
    }
    __syncthreads();
    // y = h @ W_out + b_out   (8 column tiles, one per wave, K=512)
    {
      int nb = wave * 16;
      int m0 = (lane >> 4) * 8, nn = lane & 15;
      float bv = (nb + nn) < 123 ? b_out[nb + nn] : 0.0f;
      v8f acc;
#pragma unroll
      for (int r = 0; r < 8; ++r) acc[r] = bv;
      for (int kk = 0; kk < 16; ++kk) {
        int k0 = kk * 32;
        v16bf a  = load_A_lds(&s_h[0][0], DECH, k0, lane);
        v16bf bm = load_B_gl(Wout, 128, k0, nb, lane);
        acc = __builtin_amdgcn_wmma_f32_16x16x32_bf16(false, a, false, bm, (short)0, acc, false, false);
      }
#pragma unroll
      for (int r = 0; r < 8; ++r) s_y[m0 + r][nb + nn] = acc[r];
    }
    __syncthreads();
    // GMM head: 16 lanes per batch row, lane s owns mixture s (and 16+s for s<4)
    {
      int b = tid >> 4, s = tid & 15;
      int gb = tile * 16 + b;
      int i = t + 1;
      const float* xr = x + (size_t)i * (BATCH * 5) + (size_t)gb * 5;
      float dxt = xr[0], dyt = xr[1];
      float ph0 = s_y[b][6 * s];
      float ph1 = (s < 4) ? s_y[b][6 * (16 + s)] : -1e30f;
      float lmax = fmaxf(ph0, ph1);
#pragma unroll
      for (int off = 8; off >= 1; off >>= 1) lmax = fmaxf(lmax, __shfl_xor(lmax, off, 16));
      float e0 = expf(ph0 - lmax);
      float e1 = (s < 4) ? expf(ph1 - lmax) : 0.0f;
      float esum = e0 + e1;
#pragma unroll
      for (int off = 8; off >= 1; off >>= 1) esum += __shfl_xor(esum, off, 16);
      float invden = 1.0f / esum;
      float gmm = 0.0f, smx = 0.0f, smy = 0.0f;
#pragma unroll
      for (int half = 0; half < 2; ++half) {
        int m = half ? (16 + s) : s;
        float pi = (half ? e1 : e0) * invden;
        if (!half || s < 4) {
          float mux = s_y[b][6 * m + 1], muy = s_y[b][6 * m + 2];
          float sx  = expf(s_y[b][6 * m + 3]), sy = expf(s_y[b][6 * m + 4]);
          float rho = tanhf(s_y[b][6 * m + 5]);
          float zx = (dxt - mux) / sx, zy = (dyt - muy) / sy;
          float om = 1.0f - rho * rho;
          float pdf = expf(-(zx * zx + zy * zy - 2.0f * rho * zx * zy) / (2.0f * om)) /
                      (6.2831853071795864f * sx * sy * sqrtf(om));
          gmm += pi * pdf;
          smx += pi * mux;
          smy += pi * muy;
        }
      }
#pragma unroll
      for (int off = 8; off >= 1; off >>= 1) {
        gmm += __shfl_xor(gmm, off, 16);
        smx += __shfl_xor(smx, off, 16);
        smy += __shfl_xor(smy, off, 16);
      }
      if (s == 0) {
        int len = N_s[gb];
        float q0 = s_y[b][120], q1 = s_y[b][121], q2 = s_y[b][122];
        float qm  = fmaxf(q0, fmaxf(q1, q2));
        float lse = qm + logf(expf(q0 - qm) + expf(q1 - qm) + expf(q2 - qm));
        pl_local -= (xr[2] * (q0 - lse) + xr[3] * (q1 - lse) + xr[4] * (q2 - lse)) * inv_norm;
        float mt = (i < len) ? 1.0f : 0.0f;
        ol_local -= mt * logf(gmm + 1e-6f) * inv_norm;
        float st[5];
        if (i > len) { st[0] = 0; st[1] = 0; st[2] = 0; st[3] = 0; st[4] = 1.0f; }
        else {
          int am = 0; float qb = q0;
          if (q1 > qb) { am = 1; qb = q1; }
          if (q2 > qb) { am = 2; }
          st[0] = smx; st[1] = smy;
          st[2] = (am == 0) ? 1.0f : 0.0f;
          st[3] = (am == 1) ? 1.0f : 0.0f;
          st[4] = (am == 2) ? 1.0f : 0.0f;
        }
        float* orow = out + (size_t)t * (BATCH * 5) + (size_t)gb * 5;
#pragma unroll
        for (int cix = 0; cix < 5; ++cix) { orow[cix] = st[cix]; s_in[b][cix] = (__bf16)st[cix]; }
      }
    }
  }
  if ((tid & 15) == 0) {
    atomicAdd(out + 512000, ol_local);   // offset_loss
    atomicAdd(out + 512001, pl_local);   // pen_loss
  }
}

// ---------------- host launch ----------------

extern "C" void kernel_launch(void* const* d_in, const int* in_sizes, int n_in,
                              void* d_out, int out_size, void* d_ws, size_t ws_size,
                              hipStream_t stream) {
  const float* x          = (const float*)d_in[0];
  const int*   N_s        = (const int*)  d_in[1];
  const float* eps        = (const float*)d_in[2];
  const float* enc_Wih_f  = (const float*)d_in[3];
  const float* enc_Whh_f  = (const float*)d_in[4];
  const float* enc_b_f    = (const float*)d_in[5];
  const float* enc_Wih_b  = (const float*)d_in[6];
  const float* enc_Whh_b  = (const float*)d_in[7];
  const float* enc_b_b    = (const float*)d_in[8];
  const float* W_mu       = (const float*)d_in[9];
  const float* b_mu       = (const float*)d_in[10];
  const float* W_logvar   = (const float*)d_in[11];
  const float* b_logvar   = (const float*)d_in[12];
  const float* W_fc_in    = (const float*)d_in[13];
  const float* b_fc_in    = (const float*)d_in[14];
  const float* dec_Wih    = (const float*)d_in[15];
  const float* dec_Whh    = (const float*)d_in[16];
  const float* dec_b      = (const float*)d_in[17];
  const float* W_out      = (const float*)d_in[18];
  const float* b_out      = (const float*)d_in[19];
  float* out = (float*)d_out;

  // workspace carve (byte offsets, 256B aligned)
  char* w = (char*)d_ws;
  size_t o = 0;
  auto carve = [&](size_t bytes) { char* p = w + o; o += (bytes + 255) & ~(size_t)255; return p; };
  __bf16* whhf_bf = (__bf16*)carve(262144 * 2);   // enc_Whh_f bf16
  __bf16* whhb_bf = (__bf16*)carve(262144 * 2);   // enc_Whh_b bf16
  __bf16* dwhh_bf = (__bf16*)carve(1048576 * 2);  // dec_Whh bf16
  __bf16* dwih_bf = (__bf16*)carve(327680 * 2);   // dec_Wih padded 160x2048
  __bf16* wout_bf = (__bf16*)carve(65536 * 2);    // W_out padded 512x128
  float*  hf      = (float*)carve(131072 * 4);    // encoder fwd final h
  float*  hb      = (float*)carve(131072 * 4);    // encoder bwd final h
  float*  zf      = (float*)carve(65536 * 4);     // z
  float*  h0f     = (float*)carve(262144 * 4);    // h0
  float*  c0f     = (float*)carve(262144 * 4);    // c0
  (void)ws_size; (void)n_in; (void)in_sizes; (void)out_size;

  k_init_out<<<(BATCH * 5 + 255) / 256, 256, 0, stream>>>(out);
  k_f32_to_bf16<<<(262144 + 255) / 256, 256, 0, stream>>>(enc_Whh_f, whhf_bf, 262144);
  k_f32_to_bf16<<<(262144 + 255) / 256, 256, 0, stream>>>(enc_Whh_b, whhb_bf, 262144);
  k_f32_to_bf16<<<(1048576 + 255) / 256, 256, 0, stream>>>(dec_Whh, dwhh_bf, 1048576);
  k_pad_wih<<<(327680 + 255) / 256, 256, 0, stream>>>(dec_Wih, dwih_bf);
  k_pad_wout<<<(65536 + 255) / 256, 256, 0, stream>>>(W_out, wout_bf);

  k_encoder<<<dim3(32, 2), 256, 0, stream>>>(x, enc_Wih_f, enc_b_f, enc_Wih_b, enc_b_b,
                                             whhf_bf, whhb_bf, hf, hb);

  k_latent<<<BATCH, 128, 0, stream>>>(hf, hb, eps, W_mu, b_mu, W_logvar, b_logvar,
                                      W_fc_in, b_fc_in,
                                      out + 512002, out + 512002 + 65536,
                                      zf, h0f, c0f);

  k_decoder<<<32, 256, 0, stream>>>(x, N_s, zf, h0f, c0f, dwih_bf, dwhh_bf, dec_b,
                                    wout_bf, b_out, out);
}
